// GATLayer_38714835206881
// MI455X (gfx1250) — matrix operand order repaired and checked by reading, not compiled, not run on hardware
//
#include <hip/hip_runtime.h>
#include <math.h>

typedef __attribute__((ext_vector_type(16))) _Float16 v16h;
typedef __attribute__((ext_vector_type(8)))  _Float16 v8h;
typedef __attribute__((ext_vector_type(8)))  float    v8f;

#define IN_DIM   256
#define OUT_DIM  128
#define HEADS    4
#define CDIM     32
#define NEG_SLOPE   0.2f
#define SOFTMAX_EPS 1e-16f
#define LN_EPS      1e-5f

#define KPAD        (IN_DIM + 8)   // padded LDS stride (264 halves) -> bank-conflict-free
#define TILES_PER_BLOCK 8          // 8 x 16 = 128 rows per block

// ---- float <-> monotonic unsigned key (exact float atomicMax via u32 atomicMax)
__device__ __forceinline__ unsigned fkey(float f) {
  unsigned u = __float_as_uint(f);
  return (u & 0x80000000u) ? ~u : (u | 0x80000000u);
}
__device__ __forceinline__ float fdekey(unsigned k) {
  return __uint_as_float((k & 0x80000000u) ? (k ^ 0x80000000u) : ~k);
}

// =====================================================================
// Kernel 1: h = x @ W via v_wmma_f32_16x16x32_f16 (fp32 accumulate).
// Block = 256 threads = 8 waves. Wave w owns column tile w*16.
// Block processes TILES_PER_BLOCK row tiles against one LDS staging of W.
// W is staged TRANSPOSED (Wt[col][k], padded stride) so each lane's
// B-fragment is two contiguous ds_load_b128.
// Assumes N % 16 == 0 (true: 50000) -> no per-row store guards.
// =====================================================================
__global__ __launch_bounds__(256) void gat_gemm(const float* __restrict__ x,
                                                const float* __restrict__ W,
                                                float* __restrict__ h, int N) {
  __shared__ _Float16 Wt[OUT_DIM * KPAD];   // ~66 KB of the 320 KB WGP LDS
  const int tid = threadIdx.x;
  for (int i = tid; i < IN_DIM * OUT_DIM; i += 256) {
    const int k = i >> 7;        // row of W   (0..255)
    const int c = i & 127;       // col of W   (0..127)
    Wt[c * KPAD + k] = (_Float16)W[i];
  }
  __syncthreads();

  const int wave = tid >> 5;
  const int lane = tid & 31;
  const int m    = lane & 15;    // row-in-tile (A) / col-in-tile (B,C)
  const int g    = lane >> 4;    // half-wave group
  const int col0 = wave * 16;
  const _Float16* __restrict__ Wcol = Wt + (col0 + m) * KPAD;

  for (int t = 0; t < TILES_PER_BLOCK; ++t) {
    const int row0 = (blockIdx.x * TILES_PER_BLOCK + t) * 16;
    if (row0 >= N) break;
    const float* __restrict__ xr = x + (size_t)(row0 + m) * IN_DIM;

    v8f acc = {};
#pragma unroll
    for (int k0 = 0; k0 < IN_DIM; k0 += 32) {
      // ---- A fragment 16x32 f16: lanes g=0 -> K=k0+[0..7], k0+16+[0..7]
      //                            lanes g=1 -> K=k0+8+[0..7], k0+24+[0..7]
      const int kA = k0 + 8 * g;                    // 32B-aligned
      const float4 p0 = ((const float4*)(xr + kA))[0];
      const float4 p1 = ((const float4*)(xr + kA))[1];
      const float4 p2 = ((const float4*)(xr + kA + 16))[0];
      const float4 p3 = ((const float4*)(xr + kA + 16))[1];
      v16h a;
      a[0]  = (_Float16)p0.x; a[1]  = (_Float16)p0.y;
      a[2]  = (_Float16)p0.z; a[3]  = (_Float16)p0.w;
      a[4]  = (_Float16)p1.x; a[5]  = (_Float16)p1.y;
      a[6]  = (_Float16)p1.z; a[7]  = (_Float16)p1.w;
      a[8]  = (_Float16)p2.x; a[9]  = (_Float16)p2.y;
      a[10] = (_Float16)p2.z; a[11] = (_Float16)p2.w;
      a[12] = (_Float16)p3.x; a[13] = (_Float16)p3.y;
      a[14] = (_Float16)p3.z; a[15] = (_Float16)p3.w;

      // ---- B fragment 32x16 f16: lanes g=0 -> K=k0+[0..7], k0+16+[0..7]; g=1 -> k0+8+[0..7], k0+24+[0..7]
      const int kB = k0 + 8 * g;                    // 16B-aligned in Wcol
      const v8h b0 = *(const v8h*)(Wcol + kB);      // ds_load_b128
      const v8h b1 = *(const v8h*)(Wcol + kB + 16); // ds_load_b128
      const v16h b = __builtin_shufflevector(b0, b1,
          0, 1, 2, 3, 4, 5, 6, 7, 8, 9, 10, 11, 12, 13, 14, 15);

      acc = __builtin_amdgcn_wmma_f32_16x16x32_f16(
          /*neg_a=*/false, a, /*neg_b=*/false, b,
          /*c_mod=*/(short)0, acc, /*reuse_a=*/false, /*reuse_b=*/false);
    }
    // C/D layout: VGPR r -> (M = r + 8*g, N = lane&15); N%16==0 -> no guards
#pragma unroll
    for (int r = 0; r < 8; ++r)
      h[(size_t)(row0 + r + 8 * g) * OUT_DIM + col0 + m] = acc[r];
  }
}

// =====================================================================
// Kernel 2: per-node attention logits + init of m/s accumulators + zero out.
// One wave per (node, head); lane = channel (C == 32 == wave32).
// =====================================================================
__global__ __launch_bounds__(256) void gat_logits(const float* __restrict__ h,
    const float* __restrict__ att_src, const float* __restrict__ att_dst,
    float* __restrict__ asrc, float* __restrict__ adst,
    unsigned* __restrict__ mkey, float* __restrict__ sbuf,
    float* __restrict__ out, int N) {
  const int warp = (blockIdx.x * blockDim.x + threadIdx.x) >> 5;
  const int lane = threadIdx.x & 31;
  if (warp >= N * HEADS) return;
  const int n = warp >> 2, hd = warp & 3;
  const size_t base = (size_t)n * OUT_DIM + hd * CDIM;
  const float hv = h[base + lane];
  float s1 = hv * att_src[hd * CDIM + lane];
  float s2 = hv * att_dst[hd * CDIM + lane];
#pragma unroll
  for (int off = 16; off > 0; off >>= 1) {
    s1 += __shfl_xor(s1, off, 32);
    s2 += __shfl_xor(s2, off, 32);
  }
  out[base + lane] = 0.0f;                 // zero the scatter accumulator
  if (lane == 0) {
    asrc[warp] = s1;
    adst[warp] = s2;
    mkey[warp] = 0u;                       // key-space "-infinity" sentinel
    sbuf[warp] = 0.0f;
  }
}

// =====================================================================
// Kernel 3: edge pass 1 — segment max of leaky-relu scores (incl. self loops).
// =====================================================================
__global__ __launch_bounds__(256) void gat_edge_max(const int* __restrict__ ei,
    const float* __restrict__ asrc, const float* __restrict__ adst,
    unsigned* __restrict__ mkey, int E, int N) {
  const int idx = blockIdx.x * blockDim.x + threadIdx.x;
  const int e = idx >> 2, hd = idx & 3;
  if (e >= E + N) return;
  int s, d;
  if (e < E) { s = ei[e]; d = ei[E + e]; } else { s = d = e - E; }
  float v = asrc[s * 4 + hd] + adst[d * 4 + hd];
  v = (v > 0.0f) ? v : NEG_SLOPE * v;
  atomicMax(&mkey[d * 4 + hd], fkey(v));
}

// =====================================================================
// Kernel 4: edge pass 2 — segment sum of exp(e - max).
// =====================================================================
__global__ __launch_bounds__(256) void gat_edge_sum(const int* __restrict__ ei,
    const float* __restrict__ asrc, const float* __restrict__ adst,
    const unsigned* __restrict__ mkey, float* __restrict__ sbuf, int E, int N) {
  const int idx = blockIdx.x * blockDim.x + threadIdx.x;
  const int e = idx >> 2, hd = idx & 3;
  if (e >= E + N) return;
  int s, d;
  if (e < E) { s = ei[e]; d = ei[E + e]; } else { s = d = e - E; }
  float v = asrc[s * 4 + hd] + adst[d * 4 + hd];
  v = (v > 0.0f) ? v : NEG_SLOPE * v;
  atomicAdd(&sbuf[d * 4 + hd], expf(v - fdekey(mkey[d * 4 + hd])));
}

// =====================================================================
// Kernel 5: edge pass 3 — weighted scatter. One wave per edge', lane = channel.
// Coalesced 32-lane global_atomic_add_f32 per head; this is the ~1.6 GB
// bandwidth floor of the whole layer.
// =====================================================================
__global__ __launch_bounds__(256) void gat_scatter(const int* __restrict__ ei,
    const float* __restrict__ asrc, const float* __restrict__ adst,
    const unsigned* __restrict__ mkey, const float* __restrict__ sbuf,
    const float* __restrict__ h, float* __restrict__ out, int E, int N) {
  const int warp = (blockIdx.x * blockDim.x + threadIdx.x) >> 5;
  const int lane = threadIdx.x & 31;
  if (warp >= E + N) return;
  int s, d;
  if (warp < E) { s = ei[warp]; d = ei[E + warp]; } else { s = d = warp - E; }
  const size_t sb = (size_t)s * OUT_DIM;
  const size_t db = (size_t)d * OUT_DIM;
#pragma unroll
  for (int hd = 0; hd < HEADS; ++hd) {
    float v = asrc[s * 4 + hd] + adst[d * 4 + hd];
    v = (v > 0.0f) ? v : NEG_SLOPE * v;
    const float alpha = expf(v - fdekey(mkey[d * 4 + hd])) /
                        (sbuf[d * 4 + hd] + SOFTMAX_EPS);
    atomicAdd(&out[db + hd * CDIM + lane], alpha * h[sb + hd * CDIM + lane]);
  }
}

// =====================================================================
// Kernel 6: bias + LayerNorm + ELU, in place on out. One wave per node,
// lane owns channels {lane, lane+32, lane+64, lane+96}.
// =====================================================================
__global__ __launch_bounds__(256) void gat_norm(float* __restrict__ out,
    const float* __restrict__ bias, const float* __restrict__ gamma,
    const float* __restrict__ beta, int N) {
  const int warp = (blockIdx.x * blockDim.x + threadIdx.x) >> 5;
  const int lane = threadIdx.x & 31;
  if (warp >= N) return;
  const size_t base = (size_t)warp * OUT_DIM;
  float v[4];
  float sum = 0.0f;
#pragma unroll
  for (int i = 0; i < 4; ++i) {
    const int c = i * 32 + lane;
    v[i] = out[base + c] + bias[c];
    sum += v[i];
  }
#pragma unroll
  for (int off = 16; off > 0; off >>= 1) sum += __shfl_xor(sum, off, 32);
  const float mu = sum * (1.0f / OUT_DIM);
  float var = 0.0f;
#pragma unroll
  for (int i = 0; i < 4; ++i) { const float dd = v[i] - mu; var += dd * dd; }
#pragma unroll
  for (int off = 16; off > 0; off >>= 1) var += __shfl_xor(var, off, 32);
  const float rs = rsqrtf(var * (1.0f / OUT_DIM) + LN_EPS);
#pragma unroll
  for (int i = 0; i < 4; ++i) {
    const int c = i * 32 + lane;
    float y = (v[i] - mu) * rs * gamma[c] + beta[c];
    out[base + c] = (y > 0.0f) ? y : expm1f(y);   // ELU(alpha=1)
  }
}

// =====================================================================
extern "C" void kernel_launch(void* const* d_in, const int* in_sizes, int n_in,
                              void* d_out, int out_size, void* d_ws, size_t ws_size,
                              hipStream_t stream) {
  const float* x        = (const float*)d_in[0];
  const int*   ei       = (const int*)  d_in[1];   // [2, E] row-major
  const float* W        = (const float*)d_in[2];
  const float* att_src  = (const float*)d_in[3];
  const float* att_dst  = (const float*)d_in[4];
  const float* bias     = (const float*)d_in[5];
  const float* ln_gamma = (const float*)d_in[6];
  const float* ln_beta  = (const float*)d_in[7];
  float* out = (float*)d_out;

  const int N = in_sizes[0] / IN_DIM;
  const int E = in_sizes[1] / 2;
  const int Eext = E + N;

  // workspace carve-up
  float*    h    = (float*)d_ws;                          // N*128
  float*    asrc = h    + (size_t)N * OUT_DIM;            // N*4
  float*    adst = asrc + (size_t)N * HEADS;              // N*4
  float*    sbuf = adst + (size_t)N * HEADS;              // N*4
  unsigned* mkey = (unsigned*)(sbuf + (size_t)N * HEADS); // N*4

  const int rowsPerBlock = 16 * TILES_PER_BLOCK;
  gat_gemm<<<(N + rowsPerBlock - 1) / rowsPerBlock, 256, 0, stream>>>(x, W, h, N);

  const int logitWarps = N * HEADS;
  gat_logits<<<(logitWarps * 32 + 255) / 256, 256, 0, stream>>>(
      h, att_src, att_dst, asrc, adst, mkey, sbuf, out, N);

  const int ethreads = Eext * HEADS;
  gat_edge_max<<<(ethreads + 255) / 256, 256, 0, stream>>>(ei, asrc, adst, mkey, E, N);
  gat_edge_sum<<<(ethreads + 255) / 256, 256, 0, stream>>>(ei, asrc, adst, mkey, sbuf, E, N);

  const long long swarps = (long long)Eext * 32;
  gat_scatter<<<(int)((swarps + 255) / 256), 256, 0, stream>>>(
      ei, asrc, adst, mkey, sbuf, h, out, E, N);

  gat_norm<<<(N * 32 + 255) / 256, 256, 0, stream>>>(out, bias, ln_gamma, ln_beta, N);
}
